// TopologicalQuantumAttention_54039278518599
// MI455X (gfx1250) — matrix operand order repaired and checked by reading, not compiled, not run on hardware
//
#include <hip/hip_runtime.h>
#include <hip/hip_bf16.h>
#include <math.h>

// Problem constants
#define BB   4
#define NN   2048
#define DIMC 256
#define NH   8
#define HDIM 32
// scores scale: qm.km = 2*(cos.cos+sin.sin); /sqrt(HD=32) -> 2/sqrt(32)
#define SCORE_SCALE 0.35355339059327379f

typedef __attribute__((ext_vector_type(16))) _Float16 v16h;
typedef __attribute__((ext_vector_type(8)))  float    v8f;

// Load a 16-half WMMA operand from two 16-byte chunks.
__device__ __forceinline__ v16h load2h(const _Float16* p0, const _Float16* p1) {
  union { v16h v; float4 q[2]; } u;
  u.q[0] = *(const float4*)p0;
  u.q[1] = *(const float4*)p1;
  return u.v;
}

// ---------------------------------------------------------------------------
// Pack kernels: x -> f16, W -> f16 transposed (Wt[c*256+k] = W[k*256+c])
// ---------------------------------------------------------------------------
__global__ void cvt_x_kernel(const float* __restrict__ x, _Float16* __restrict__ xh) {
  int i = blockIdx.x * 256 + threadIdx.x;          // 0 .. B*N*DIM-1
  xh[i] = (_Float16)x[i];
}

__global__ void cvt_w_kernel(const float* __restrict__ Wq, const float* __restrict__ Wk,
                             const float* __restrict__ Wv, const float* __restrict__ Wo,
                             _Float16* __restrict__ Wt) {
  int m = blockIdx.z;
  const float* W = (m == 0) ? Wq : (m == 1) ? Wk : (m == 2) ? Wv : Wo;
  int i = blockIdx.x * 256 + threadIdx.x;          // 0 .. 65535, i = c*256 + k
  int c = i >> 8, k = i & 255;
  Wt[m * 65536 + i] = (_Float16)W[k * 256 + c];
}

// ---------------------------------------------------------------------------
// Shared GEMM tile routine: one wave computes rows [rowbase,rowbase+16) x
// cols [c0,c0+64) of A[8192x256] @ B[256x256], f16 in / f32 accumulate.
// A row-major (ld=256), Bt packed column-major (Bt[c*256+k]).
// ---------------------------------------------------------------------------
__device__ __forceinline__ void gemm16x64(const _Float16* __restrict__ A,
                                          const _Float16* __restrict__ Bt,
                                          long rowbase, int c0, int lane, v8f acc[4]) {
  const int l15 = lane & 15, lhi = lane >> 4;
#pragma unroll
  for (int kc = 0; kc < 8; ++kc) {
    const int k0 = kc * 32;
    const _Float16* ap = A + (rowbase + l15) * DIMC + k0 + 16 * lhi;
    v16h a = load2h(ap, ap + 8);
#pragma unroll
    for (int jt = 0; jt < 4; ++jt) {
      const _Float16* bp = Bt + (long)(c0 + jt * 16 + l15) * DIMC + k0 + 16 * lhi;
      v16h b = load2h(bp, bp + 8);
      acc[jt] = __builtin_amdgcn_wmma_f32_16x16x32_f16(false, a, false, b,
                                                       (short)0, acc[jt], false, false);
    }
  }
}

// ---------------------------------------------------------------------------
// QKV projection + quantum feature map.
//   qf/kf: [row(=b*N+n)][h][64]  (cos in d, sin in 32+d), f16
//   vt:    [b][h][d][n]          (transposed for B-operand loads), f16
// grid (64, 4, 3), block 256 (8 waves, one 16-row tile each)
// ---------------------------------------------------------------------------
__global__ void __launch_bounds__(256)
qkv_kernel(const _Float16* __restrict__ xh, const _Float16* __restrict__ Wt,
           const float* __restrict__ bq, const float* __restrict__ bk,
           const float* __restrict__ bv,
           _Float16* __restrict__ qf, _Float16* __restrict__ kf,
           _Float16* __restrict__ vt) {
  const int lane = threadIdx.x & 31, wave = threadIdx.x >> 5;
  const int l15 = lane & 15, lhi = lane >> 4;
  const long rowbase = (long)(blockIdx.x * 8 + wave) * 16;
  const int c0 = blockIdx.y * 64;
  const int mat = blockIdx.z;

  const _Float16* Bt = Wt + (long)mat * 65536;
  const float* bias = (mat == 0) ? bq : (mat == 1) ? bk : bv;

  v8f acc[4] = {};
  gemm16x64(xh, Bt, rowbase, c0, lane, acc);

#pragma unroll
  for (int jt = 0; jt < 4; ++jt) {
    const int c = c0 + jt * 16 + l15;
    const float bval = bias[c];
    const int h = c >> 5, d = c & 31;
#pragma unroll
    for (int r = 0; r < 8; ++r) {
      const long row = rowbase + r + 8 * lhi;
      const float v = acc[jt][r] + bval;
      if (mat < 2) {
        _Float16* dst = ((mat == 0) ? qf : kf) + (row * NH + h) * 64;
        float s, cns;
        __sincosf(v, &s, &cns);
        dst[d]      = (_Float16)cns;
        dst[32 + d] = (_Float16)s;
      } else {
        const int b = (int)(row >> 11);
        const int n = (int)(row & (NN - 1));
        vt[((long)(b * NH + h) * HDIM + d) * NN + n] = (_Float16)v;
      }
    }
  }
}

// ---------------------------------------------------------------------------
// Flash attention with topology mask. One wave owns a 16-row query tile of
// one (b,h). m processed in chunks of 32: 4 WMMA for S = qf @ kf^T (K=64),
// topo mask + online softmax, P staged through LDS into A-layout,
// 2 WMMA for O += P @ V (K=32).
// grid (B*H = 32, 16), block 256 (8 waves).
// ---------------------------------------------------------------------------
__global__ void __launch_bounds__(256)
attn_kernel(const _Float16* __restrict__ qf, const _Float16* __restrict__ kf,
            const _Float16* __restrict__ vt, const float* __restrict__ topo,
            _Float16* __restrict__ ao) {
  __shared__ __align__(16) _Float16 P[8][16 * 32];

  const int lane = threadIdx.x & 31, wave = threadIdx.x >> 5;
  const int l15 = lane & 15, lhi = lane >> 4;
  const int bh = blockIdx.x;
  const int b = bh >> 3, h = bh & 7;
  const int n0 = (blockIdx.y * 8 + wave) * 16;
  const long rowbase = (long)b * NN + n0;

  // Q operands (K = feature dim 64 -> two 16x32 A operands), resident in VGPRs
  const _Float16* qrow = qf + ((rowbase + l15) * NH + h) * 64;
  const v16h aq0 = load2h(qrow + 16 * lhi,      qrow + 16 * lhi + 8);
  const v16h aq1 = load2h(qrow + 32 + 16 * lhi, qrow + 32 + 16 * lhi + 8);

  v8f o0 = {}, o1 = {};
  float rmax[8], rsum[8];
#pragma unroll
  for (int r = 0; r < 8; ++r) { rmax[r] = -INFINITY; rsum[r] = 0.0f; }

  const long vbase0 = ((long)(b * NH + h) * HDIM + l15) * NN;        // d = l15
  const long vbase1 = ((long)(b * NH + h) * HDIM + 16 + l15) * NN;   // d = 16+l15

  for (int m0 = 0; m0 < NN; m0 += 32) {
    // Prefetch next topo tile (16 rows x 32 cols) into cache hierarchy.
    if (m0 + 32 < NN) {
      const float* pf = topo + ((long)b * NN + n0 + (lane >> 1)) * NN
                             + (m0 + 32) + (lane & 1) * 16;
      __builtin_prefetch(pf, 0, 0);
    }

    // ---- S = qf @ kf^T for this 16x32 tile (two 16x16 C tiles) ----
    const _Float16* k0p = kf + (((long)b * NN + m0 + l15) * NH + h) * 64;
    const _Float16* k1p = kf + (((long)b * NN + m0 + 16 + l15) * NH + h) * 64;
    v8f s0 = {}, s1 = {};
    {
      v16h b00 = load2h(k0p + 16 * lhi,      k0p + 16 * lhi + 8);   // d 0..31
      v16h b01 = load2h(k0p + 32 + 16 * lhi, k0p + 32 + 16 * lhi + 8); // d 32..63
      v16h b10 = load2h(k1p + 16 * lhi,      k1p + 16 * lhi + 8);
      v16h b11 = load2h(k1p + 32 + 16 * lhi, k1p + 32 + 16 * lhi + 8);
      s0 = __builtin_amdgcn_wmma_f32_16x16x32_f16(false, aq0, false, b00, (short)0, s0, false, false);
      s0 = __builtin_amdgcn_wmma_f32_16x16x32_f16(false, aq1, false, b01, (short)0, s0, false, false);
      s1 = __builtin_amdgcn_wmma_f32_16x16x32_f16(false, aq0, false, b10, (short)0, s1, false, false);
      s1 = __builtin_amdgcn_wmma_f32_16x16x32_f16(false, aq1, false, b11, (short)0, s1, false, false);
    }

    // ---- topo mask + scale + online softmax (per C-layout row r) ----
#pragma unroll
    for (int r = 0; r < 8; ++r) {
      const float* tp = topo + ((long)b * NN + n0 + r + 8 * lhi) * NN + m0;
      float v0 = s0[r] * SCORE_SCALE * tp[l15];
      float v1 = s1[r] * SCORE_SCALE * tp[16 + l15];

      float mx = fmaxf(v0, v1);
      mx = fmaxf(mx, __shfl_xor(mx, 1));
      mx = fmaxf(mx, __shfl_xor(mx, 2));
      mx = fmaxf(mx, __shfl_xor(mx, 4));
      mx = fmaxf(mx, __shfl_xor(mx, 8));

      const float newm = fmaxf(rmax[r], mx);
      const float cf = __expf(rmax[r] - newm);
      rmax[r] = newm;
      o0[r] *= cf;
      o1[r] *= cf;

      const float e0 = __expf(v0 - newm);
      const float e1 = __expf(v1 - newm);
      float ps = e0 + e1;
      ps += __shfl_xor(ps, 1);
      ps += __shfl_xor(ps, 2);
      ps += __shfl_xor(ps, 4);
      ps += __shfl_xor(ps, 8);
      rsum[r] = rsum[r] * cf + ps;

      // Stage P (C layout -> row-major LDS)
      P[wave][(r + 8 * lhi) * 32 + l15]      = (_Float16)e0;
      P[wave][(r + 8 * lhi) * 32 + 16 + l15] = (_Float16)e1;
    }

    // Same-wave LDS in-order guarantees data; just block compiler reordering.
    __asm volatile("" ::: "memory");
    __builtin_amdgcn_sched_barrier(0);

    // ---- O += P @ V (K = 32 m values) ----
    const _Float16* pr = &P[wave][l15 * 32 + 16 * lhi];
    v16h ap = load2h(pr, pr + 8);
    const _Float16* v0p = vt + vbase0 + m0 + 16 * lhi;
    const _Float16* v1p = vt + vbase1 + m0 + 16 * lhi;
    v16h bv0 = load2h(v0p, v0p + 8);
    v16h bv1 = load2h(v1p, v1p + 8);
    o0 = __builtin_amdgcn_wmma_f32_16x16x32_f16(false, ap, false, bv0, (short)0, o0, false, false);
    o1 = __builtin_amdgcn_wmma_f32_16x16x32_f16(false, ap, false, bv1, (short)0, o1, false, false);

    __builtin_amdgcn_sched_barrier(0);
    __asm volatile("" ::: "memory");
  }

  // ---- normalize + store f16 [row][h*32 + d] for the output projection ----
#pragma unroll
  for (int r = 0; r < 8; ++r) {
    const float inv = 1.0f / rsum[r];
    _Float16* dst = ao + (rowbase + r + 8 * lhi) * DIMC + h * HDIM;
    dst[l15]      = (_Float16)(o0[r] * inv);
    dst[16 + l15] = (_Float16)(o1[r] * inv);
  }
}

// ---------------------------------------------------------------------------
// Output projection: ao[8192x256] @ Wo + bo -> f32 out
// grid (64, 4), block 256.
// ---------------------------------------------------------------------------
__global__ void __launch_bounds__(256)
outproj_kernel(const _Float16* __restrict__ ao, const _Float16* __restrict__ Wot,
               const float* __restrict__ bo, float* __restrict__ out) {
  const int lane = threadIdx.x & 31, wave = threadIdx.x >> 5;
  const int l15 = lane & 15, lhi = lane >> 4;
  const long rowbase = (long)(blockIdx.x * 8 + wave) * 16;
  const int c0 = blockIdx.y * 64;

  v8f acc[4] = {};
  gemm16x64(ao, Wot, rowbase, c0, lane, acc);

#pragma unroll
  for (int jt = 0; jt < 4; ++jt) {
    const int c = c0 + jt * 16 + l15;
    const float bval = bo[c];
#pragma unroll
    for (int r = 0; r < 8; ++r)
      out[(rowbase + r + 8 * lhi) * DIMC + c] = acc[jt][r] + bval;
  }
}

// ---------------------------------------------------------------------------
extern "C" void kernel_launch(void* const* d_in, const int* in_sizes, int n_in,
                              void* d_out, int out_size, void* d_ws, size_t ws_size,
                              hipStream_t stream) {
  (void)in_sizes; (void)n_in; (void)out_size; (void)ws_size;
  const float* x    = (const float*)d_in[0];
  const float* topo = (const float*)d_in[1];
  const float* Wq   = (const float*)d_in[2];
  const float* bq   = (const float*)d_in[3];
  const float* Wk   = (const float*)d_in[4];
  const float* bk   = (const float*)d_in[5];
  const float* Wv   = (const float*)d_in[6];
  const float* bv   = (const float*)d_in[7];
  const float* Wo   = (const float*)d_in[8];
  const float* bo   = (const float*)d_in[9];
  float* out = (float*)d_out;

  char* ws = (char*)d_ws;
  _Float16* xh = (_Float16*)(ws);                                       //  4 MB
  _Float16* Wt = (_Float16*)(ws + (4u << 20));                          // 512 KB (4 mats)
  _Float16* qf = (_Float16*)(ws + (4u << 20) + (512u << 10));           //  8 MB
  _Float16* kf = (_Float16*)(ws + (12u << 20) + (512u << 10));          //  8 MB
  _Float16* vt = (_Float16*)(ws + (20u << 20) + (512u << 10));          //  4 MB
  _Float16* ao = (_Float16*)(ws + (24u << 20) + (512u << 10));          //  4 MB

  cvt_x_kernel<<<dim3(BB * NN * DIMC / 256), dim3(256), 0, stream>>>(x, xh);
  cvt_w_kernel<<<dim3(256, 1, 4), dim3(256), 0, stream>>>(Wq, Wk, Wv, Wo, Wt);
  qkv_kernel<<<dim3(64, 4, 3), dim3(256), 0, stream>>>(xh, Wt, bq, bk, bv, qf, kf, vt);
  attn_kernel<<<dim3(BB * NH, NN / 128), dim3(256), 0, stream>>>(qf, kf, vt, topo, ao);
  outproj_kernel<<<dim3(64, 4), dim3(256), 0, stream>>>(ao, Wt + 3 * 65536, bo, out);
}
